// HeteroGNNModel_87333864997150
// MI455X (gfx1250) — matrix operand order repaired and checked by reading, not compiled, not run on hardware
//
#include <hip/hip_runtime.h>
#include <hip/hip_bf16.h>

// ---------------------------------------------------------------------------
// HeteroGNN (GINEConv x2) for MI455X / gfx1250.
// All dense math ([M,64]x[64,64] GEMMs) runs through v_wmma_f32_16x16x32_bf16
// (bf16 inputs, f32 accumulate). Edge gather + ReLU + scatter-add is fused
// into the edge-lin WMMA kernel so each 64-wide edge vector crosses HBM once.
// Encoded edge features are cached in bf16 (halves the 2x re-read per layer).
// A-matrix fragments are hoisted out of the N-tile loop: one 16x64 A slice
// feeds all four 16x16 output tiles (2 fragment loads instead of 8).
//
// Workspace layout (carved from d_ws, ~742 MB):
//   xC/xH/xO     : layer-input node features, f32 [N,64]
//   yC/yH/yO     : per-layer accumulation of relation MLP outputs, f32 [N,64]
//   agg          : per-relation message aggregation buffer, f32 [300k,64]
//   eaEnc[0..5]  : encoded edge features, bf16 [400k,64]
// ---------------------------------------------------------------------------

typedef __bf16 bf16_t;
typedef __attribute__((ext_vector_type(16))) __bf16 v16bf;
typedef __attribute__((ext_vector_type(8)))  __bf16 v8bf;
typedef __attribute__((ext_vector_type(8)))  float  v8f;

// --- WMMA fragment loaders (layouts per CDNA5 ISA 7.12.2) -------------------
// A (16x32 bf16): lanes 0-15 hold row M=lane, K = {0..7, 16..23};
//                 lanes 16-31 hold row M=lane-16, K = {8..15, 24..31}.
// rowptr points at the start of this lane's 64-element row (row-major, stride 64).
__device__ inline v16bf load_a_frag_row(const bf16_t* rowptr, int lane, int kk) {
    const int base0 = (lane >> 4) * 8;
    v8bf x0 = *reinterpret_cast<const v8bf*>(rowptr + kk + base0);
    v8bf x1 = *reinterpret_cast<const v8bf*>(rowptr + kk + base0 + 16);
    v16bf a;
#pragma unroll
    for (int j = 0; j < 8; ++j) { a[j] = x0[j]; a[8 + j] = x1[j]; }
    return a;
}

// B (32x16 bf16): lanes 0-15 hold col N=lane, K = kk+{0..7, 16..23};
//                 lanes 16-31 hold col N=lane-16, K = kk+{8..15, 24..31}.
// bt is the weight stored transposed in LDS: bt[n*64 + k] = W[k][n].
__device__ inline v16bf load_b_frag(const bf16_t* bt, int lane, int ntile, int kk) {
    const int n  = (lane & 15) + ntile * 16;
    const int kb = kk + (lane >> 4) * 8;
    const bf16_t* p = bt + n * 64 + kb;
    v8bf x0 = *reinterpret_cast<const v8bf*>(p);
    v8bf x1 = *reinterpret_cast<const v8bf*>(p + 16);
    v16bf b;
#pragma unroll
    for (int j = 0; j < 8; ++j) { b[j] = x0[j]; b[8 + j] = x1[j]; }
    return b;
}

// 16x64 @ 64x64 -> four 16x16 f32 tiles. A fragments hoisted: loaded once,
// reused across all four N-tiles.
__device__ inline void gemm_16x64x64(const bf16_t* arow, const bf16_t* bt,
                                     int lane, v8f acc[4]) {
    const v16bf a0 = load_a_frag_row(arow, lane, 0);
    const v16bf a1 = load_a_frag_row(arow, lane, 32);
#pragma unroll
    for (int nt = 0; nt < 4; ++nt) {
        v8f c = {};
        c = __builtin_amdgcn_wmma_f32_16x16x32_bf16(
                false, a0, false, load_b_frag(bt, lane, nt, 0),
                (short)0, c, false, false);
        c = __builtin_amdgcn_wmma_f32_16x16x32_bf16(
                false, a1, false, load_b_frag(bt, lane, nt, 32),
                (short)0, c, false, false);
        acc[nt] = c;
    }
}

// ---------------------------------------------------------------------------
// Fused 2-layer MLP:  Y = (relu(Xin @ W1 + b1)) @ W2 + b2
// where Xin = scale * X + X2   (scale = 1 + *epsPtr, X2/agg optional).
// Block = 128 threads = 4 waves; each wave computes a 16x64 output tile.
// out_kind: 0 = f32 store, 1 = f32 accumulate, 2 = bf16 store.
// ---------------------------------------------------------------------------
__global__ __launch_bounds__(128) void mlp2_wmma_kernel(
    const float* __restrict__ X, int M, int Kin,
    const float* __restrict__ X2,       // optional [M,64]
    const float* __restrict__ epsPtr,   // optional scale = 1 + eps
    const float* __restrict__ W1, const float* __restrict__ B1,
    const float* __restrict__ W2, const float* __restrict__ B2,
    void* __restrict__ Yout, int out_kind, int relu_out)
{
    __shared__ __align__(16) bf16_t w1t[64 * 64];      // W1 transposed+padded
    __shared__ __align__(16) bf16_t w2t[64 * 64];      // W2 transposed
    __shared__ __align__(16) bf16_t aw[4][16 * 64];    // per-wave input tile
    __shared__ __align__(16) bf16_t hw[4][16 * 64];    // per-wave hidden tile

    const int tid  = threadIdx.x;
    const int lane = tid & 31;
    const int wave = tid >> 5;

    // Stage weights transposed (bt[n][k] = W[k][n]), pad K to 64 with zeros.
    for (int i = tid; i < 64 * 64; i += 128) {
        const int n = i >> 6, k = i & 63;
        w1t[i] = (k < Kin) ? (bf16_t)W1[k * 64 + n] : (bf16_t)0.0f;
        w2t[i] = (bf16_t)W2[k * 64 + n];
    }
    __syncthreads();

    const float scale   = 1.0f + (epsPtr ? epsPtr[0] : 0.0f);
    const int   rowbase = blockIdx.x * 64 + wave * 16;

    // Stage this wave's 16 input rows into LDS as bf16 (zero padded).
    bf16_t* atile = aw[wave];
    for (int i = lane; i < 16 * 64; i += 32) {
        const int r = i >> 6, k = i & 63;
        const int row = rowbase + r;
        float v = 0.0f;
        if (row < M) {
            if (k < Kin) v = scale * X[(size_t)row * Kin + k];
            if (X2)      v += X2[(size_t)row * 64 + k];
        }
        atile[i] = (bf16_t)v;
    }

    // GEMM 1: hidden = relu(Xin @ W1 + b1), written bf16 into hw[wave].
    bf16_t* htile = hw[wave];
    {
        v8f acc[4];
        gemm_16x64x64(atile + (lane & 15) * 64, w1t, lane, acc);
        const int r0 = (lane >> 4) * 8;
#pragma unroll
        for (int nt = 0; nt < 4; ++nt) {
            const int n = nt * 16 + (lane & 15);
            const float bias = B1[n];
#pragma unroll
            for (int t = 0; t < 8; ++t) {
                float h = acc[nt][t] + bias;
                h = h > 0.0f ? h : 0.0f;
                htile[(r0 + t) * 64 + n] = (bf16_t)h;
            }
        }
    }

    // GEMM 2: out = hidden @ W2 + b2.
    {
        v8f acc[4];
        gemm_16x64x64(htile + (lane & 15) * 64, w2t, lane, acc);
        const int r0 = (lane >> 4) * 8;
#pragma unroll
        for (int nt = 0; nt < 4; ++nt) {
            const int n = nt * 16 + (lane & 15);
            const float bias = B2[n];
#pragma unroll
            for (int t = 0; t < 8; ++t) {
                const int row = rowbase + r0 + t;
                if (row >= M) continue;
                float o = acc[nt][t] + bias;
                if (relu_out) o = o > 0.0f ? o : 0.0f;
                const size_t idx = (size_t)row * 64 + n;
                if (out_kind == 0)      ((float*)Yout)[idx]  = o;
                else if (out_kind == 1) ((float*)Yout)[idx] += o;
                else                    ((bf16_t*)Yout)[idx] = (bf16_t)o;
            }
        }
    }
}

// ---------------------------------------------------------------------------
// Fused GINE message kernel for one relation:
//   msg = relu( x_src[src[e]] + eaEnc[e] @ lW + lb );  agg[dst[e]] += msg
// A fragments are loaded straight from global bf16 (row-major, two 16B
// chunks per lane, loaded once for all four N-tiles); epilogue gathers x_src
// and scatter-adds with f32 atomics.
// ---------------------------------------------------------------------------
__global__ __launch_bounds__(128) void gine_msg_kernel(
    const bf16_t* __restrict__ eaEnc,  // [E,64] bf16
    const float*  __restrict__ lW,     // [64,64]
    const float*  __restrict__ lb,     // [64]
    const int*    __restrict__ ei,     // [2,E]: row0 = src, row1 = dst
    const float*  __restrict__ xsrc,   // [Ns,64]
    float*        __restrict__ agg,    // [Nd,64]
    int E)
{
    __shared__ __align__(16) bf16_t wt[64 * 64];
    const int tid  = threadIdx.x;
    const int lane = tid & 31;
    const int wave = tid >> 5;

    for (int i = tid; i < 64 * 64; i += 128) {
        const int n = i >> 6, k = i & 63;
        wt[i] = (bf16_t)lW[k * 64 + n];
    }
    __syncthreads();

    const int ebase = blockIdx.x * 64 + wave * 16;
    if (ebase >= E) return;

    // This lane's A row = edge (ebase + lane&15), clamped; OOB rows are
    // masked in the epilogue.
    int er = ebase + (lane & 15);
    if (er >= E) er = E - 1;

    v8f acc[4];
    gemm_16x64x64(eaEnc + (size_t)er * 64, wt, lane, acc);

    const int r0 = (lane >> 4) * 8;
#pragma unroll
    for (int nt = 0; nt < 4; ++nt) {
        const int n = nt * 16 + (lane & 15);
        const float bias = lb[n];
#pragma unroll
        for (int t = 0; t < 8; ++t) {
            const int e = ebase + r0 + t;
            if (e >= E) continue;
            const int s = ei[e];       // src node
            const int d = ei[E + e];   // dst node
            float m = acc[nt][t] + bias + xsrc[(size_t)s * 64 + n];
            m = m > 0.0f ? m : 0.0f;
            atomicAdd(&agg[(size_t)d * 64 + n], m);
        }
    }
}

__global__ void zero_kernel(float* __restrict__ p, size_t n) {
    const size_t i = (size_t)blockIdx.x * blockDim.x + threadIdx.x;
    if (i < n) p[i] = 0.0f;
}

__global__ void relu_copy_kernel(float* __restrict__ x,
                                 const float* __restrict__ y, size_t n) {
    const size_t i = (size_t)blockIdx.x * blockDim.x + threadIdx.x;
    if (i < n) { const float v = y[i]; x[i] = v > 0.0f ? v : 0.0f; }
}

// out[i] = dot(x[i,0:64], w) + b   (trivial GEMV tail)
__global__ void head_kernel(const float* __restrict__ x,
                            const float* __restrict__ w,
                            const float* __restrict__ b,
                            float* __restrict__ out, int n) {
    const int i = blockIdx.x * blockDim.x + threadIdx.x;
    if (i >= n) return;
    const float* row = x + (size_t)i * 64;
    float acc = b[0];
#pragma unroll
    for (int k = 0; k < 64; ++k) acc += row[k] * w[k];
    out[i] = acc;
}

// ---------------------------------------------------------------------------
extern "C" void kernel_launch(void* const* d_in, const int* in_sizes, int n_in,
                              void* d_out, int out_size, void* d_ws, size_t ws_size,
                              hipStream_t stream) {
    (void)in_sizes; (void)n_in; (void)out_size; (void)ws_size;
    constexpr int NC = 300000, NH = 300000, NO = 100000, NE = 400000;

    const float* xCin = (const float*)d_in[0];
    const float* xHin = (const float*)d_in[1];
    const float* xOin = (const float*)d_in[2];
    const float* ea[6];
    const int*   ei[6];
    for (int r = 0; r < 6; ++r) ea[r] = (const float*)d_in[3 + r];
    for (int r = 0; r < 6; ++r) ei[r] = (const int*)d_in[9 + r];
    const float* cW1 = (const float*)d_in[15], *cb1 = (const float*)d_in[16];
    const float* cW2 = (const float*)d_in[17], *cb2 = (const float*)d_in[18];
    const float* hW1 = (const float*)d_in[19], *hb1 = (const float*)d_in[20];
    const float* hW2 = (const float*)d_in[21], *hb2 = (const float*)d_in[22];
    const float* oW1 = (const float*)d_in[23], *ob1 = (const float*)d_in[24];
    const float* oW2 = (const float*)d_in[25], *ob2 = (const float*)d_in[26];
    const float* eW1 = (const float*)d_in[27], *eb1 = (const float*)d_in[28];
    const float* eW2 = (const float*)d_in[29], *eb2 = (const float*)d_in[30];
    const float* nW1 = (const float*)d_in[31], *nb1 = (const float*)d_in[32];
    const float* nW2 = (const float*)d_in[33], *nb2 = (const float*)d_in[34];
    const float* eps = (const float*)d_in[35];
    const float* lW  = (const float*)d_in[36];
    const float* lb  = (const float*)d_in[37];
    const float* coutW = (const float*)d_in[38], *coutb = (const float*)d_in[39];
    const float* houtW = (const float*)d_in[40], *houtb = (const float*)d_in[41];

    // ---- workspace carve (256B aligned chunks) ----
    char* wp = (char*)d_ws;
    auto carve = [&](size_t bytes) -> void* {
        void* r = (void*)wp;
        wp += (bytes + 255) & ~(size_t)255;
        return r;
    };
    float* xC = (float*)carve((size_t)NC * 64 * 4);
    float* xH = (float*)carve((size_t)NH * 64 * 4);
    float* xO = (float*)carve((size_t)NO * 64 * 4);
    float* yC = (float*)carve((size_t)NC * 64 * 4);
    float* yH = (float*)carve((size_t)NH * 64 * 4);
    float* yO = (float*)carve((size_t)NO * 64 * 4);
    float* agg = (float*)carve((size_t)NC * 64 * 4);   // max dst count
    bf16_t* eaE[6];
    for (int r = 0; r < 6; ++r) eaE[r] = (bf16_t*)carve((size_t)NE * 64 * 2);

    auto mlp2 = [&](const float* X, int M, int Kin, const float* X2,
                    const float* epsP, const float* W1, const float* B1,
                    const float* W2, const float* B2, void* Y, int kind) {
        mlp2_wmma_kernel<<<(M + 63) / 64, 128, 0, stream>>>(
            X, M, Kin, X2, epsP, W1, B1, W2, B2, Y, kind, 0);
    };

    // ---- node & edge encoders ----
    mlp2(xCin, NC, 38, nullptr, nullptr, cW1, cb1, cW2, cb2, xC, 0);
    mlp2(xHin, NH, 32, nullptr, nullptr, hW1, hb1, hW2, hb2, xH, 0);
    mlp2(xOin, NO, 6,  nullptr, nullptr, oW1, ob1, oW2, ob2, xO, 0);
    for (int r = 0; r < 6; ++r)
        mlp2(ea[r], NE, 4, nullptr, nullptr, eW1, eb1, eW2, eb2, eaE[r], 2);

    // relation tables: {CC, CH, HH, CO, HO, OO}
    const float* srcX[6] = {xC, xC, xH, xC, xH, xO};
    float*       dstX[6] = {xC, xH, xH, xO, xO, xO};
    float*       dstY[6] = {yC, yH, yH, yO, yO, yO};
    const int    Ndst[6] = {NC, NH, NH, NO, NO, NO};
    const int    mode[6] = {0, 0, 1, 0, 1, 1};   // store first rel per dst, then add

    for (int l = 0; l < 2; ++l) {
        for (int r = 0; r < 6; ++r) {
            const size_t an = (size_t)Ndst[r] * 64;
            zero_kernel<<<(unsigned)((an + 255) / 256), 256, 0, stream>>>(agg, an);
            gine_msg_kernel<<<(NE + 63) / 64, 128, 0, stream>>>(
                eaE[r], lW + (size_t)(l * 6 + r) * 4096,
                lb + (size_t)(l * 6 + r) * 64, ei[r], srcX[r], agg, NE);
            mlp2(dstX[r], Ndst[r], 64, agg, eps + (l * 6 + r),
                 nW1, nb1, nW2, nb2, dstY[r], mode[r]);
        }
        const size_t nc = (size_t)NC * 64, nh = (size_t)NH * 64, no = (size_t)NO * 64;
        relu_copy_kernel<<<(unsigned)((nc + 255) / 256), 256, 0, stream>>>(xC, yC, nc);
        relu_copy_kernel<<<(unsigned)((nh + 255) / 256), 256, 0, stream>>>(xH, yH, nh);
        relu_copy_kernel<<<(unsigned)((no + 255) / 256), 256, 0, stream>>>(xO, yO, no);
    }

    // ---- output heads: (xC @ cout_W + cout_b, xH @ hout_W + hout_b) ----
    float* out = (float*)d_out;
    head_kernel<<<(NC + 255) / 256, 256, 0, stream>>>(xC, coutW, coutb, out, NC);
    head_kernel<<<(NH + 255) / 256, 256, 0, stream>>>(xH, houtW, houtb, out + NC, NH);
}
